// MoELayer_90237262889053
// MI455X (gfx1250) — matrix operand order repaired and checked by reading, compiled, not run on hardware
//
#include <hip/hip_runtime.h>
#include <hip/hip_bf16.h>
#include <math.h>

// MoE: E=8 experts, H=1024, I=4096, T=B*S=8192 tokens.
#define E_ 8
#define H_ 1024
#define I_ 4096
#define T_ 8192

typedef __attribute__((ext_vector_type(16))) _Float16 v16h;
typedef __attribute__((ext_vector_type(8)))  _Float16 v8h;
typedef __attribute__((ext_vector_type(8)))  float    v8f;
typedef __attribute__((ext_vector_type(4)))  unsigned v4u;
typedef __attribute__((ext_vector_type(8)))  int      v8i;
typedef __attribute__((ext_vector_type(4)))  int      v4i;

// ---- CDNA5 async global->LDS copy (per-lane 16B), tracked on ASYNCcnt
__device__ __forceinline__ void async_g2l_b128(unsigned lds_off, const void* gaddr) {
    unsigned long long ga = (unsigned long long)gaddr;
    asm volatile("global_load_async_to_lds_b128 %0, %1, off"
                 :: "v"(lds_off), "v"(ga) : "memory");
}
__device__ __forceinline__ void wait_async0() {
    asm volatile("s_wait_asynccnt 0" ::: "memory");
}

// ---- pack two f32 -> two f16 -> one DS b32 store
__device__ __forceinline__ void store_pair_f16(_Float16* p, float lo, float hi) {
    union { _Float16 h[2]; unsigned u; } t;
    t.h[0] = (_Float16)lo;
    t.h[1] = (_Float16)hi;
    *(unsigned*)p = t.u;
}

// ---- CDNA5 Tensor Data Mover: 2D tile (64 rows x 32 f16) global -> LDS.
// D# per CDNA5 ISA. Padding: 16B after every 64B stored
// (pad_interval=3 -> 16 DWORDs, pad_amount=3 -> 4 DWORDs) to match the
// LDS row stride of 80B ([64][40] f16). tensor_dim1=rows_rem gives free
// zero-fill for short tiles via TDM OOB handling.
__device__ __forceinline__ void tdm_load_tile_f16(unsigned lds_off, const void* gaddr,
                                                  unsigned rows_rem) {
    const unsigned long long ga = (unsigned long long)gaddr;
    const v4u g0 = { 1u,                                   // count=1, valid
                     lds_off,                              // lds_addr (bytes)
                     (unsigned)ga,                         // global_addr[31:0]
                     ((unsigned)(ga >> 32) & 0x01FFFFFFu)  // global_addr[56:32]
                         | 0x80000000u };                  // type=2 ("image")
    const unsigned w0 = (1u << 16)      // data_size = 1 (2 bytes)
                      | (1u << 20)      // pad_enable
                      | (3u << 22)      // pad_interval: 16 DWORDs
                      | (3u << 25);     // pad_amount:    4 DWORDs
    const unsigned dim0 = I_;           // tensor row length (elements)
    const unsigned dim1 = rows_rem;     // rows available from tile start
    const v8i g1 = { (int)w0,
                     (int)((dim0 & 0xFFFFu) << 16),                  // dim0[15:0]
                     (int)((dim0 >> 16) | ((dim1 & 0xFFFFu) << 16)), // dim0[31:16]|dim1[15:0]
                     (int)((dim1 >> 16) | (32u << 16)),              // dim1[31:16]|tile_dim0=32
                     (int)64u,                                       // tile_dim1=64, tile_dim2=0
                     (int)(unsigned)I_,                              // tensor_dim0_stride[31:0]
                     0, 0 };
    const v4i z4 = { 0, 0, 0, 0 };
#if __clang_major__ >= 23
    const v8i z8 = { 0, 0, 0, 0, 0, 0, 0, 0 };
    __builtin_amdgcn_tensor_load_to_lds(g0, g1, z4, z4, z8, 0);
#else
    __builtin_amdgcn_tensor_load_to_lds(g0, g1, z4, z4, 0);
#endif
}

// ---------------------------------------------------------------- init
__global__ __launch_bounds__(32) void moe_init(int* cnt) {
    if (threadIdx.x < E_) cnt[threadIdx.x] = 0;
}

// ---------------------------------------------------------------- router
__global__ __launch_bounds__(256) void moe_router(const float* __restrict__ x,
                                                  const float* __restrict__ wr,
                                                  const float* __restrict__ br,
                                                  _Float16* __restrict__ xh,
                                                  int*   __restrict__ top1,
                                                  int*   __restrict__ rank,
                                                  float* __restrict__ gate,
                                                  int*   __restrict__ cnt) {
    const int lane = threadIdx.x & 31;
    const int wave = threadIdx.x >> 5;
    const int t = blockIdx.x * 8 + wave;
    const float* xr = x + (size_t)t * H_;
    float acc[E_];
#pragma unroll
    for (int e = 0; e < E_; ++e) acc[e] = 0.f;
    for (int j = 0; j < H_ / 32; ++j) {
        const int h = j * 32 + lane;
        const float xv = xr[h];
        xh[(size_t)t * H_ + h] = (_Float16)xv;
        const float4* wp = (const float4*)(wr + (size_t)h * E_);
        const float4 w0 = wp[0], w1v = wp[1];
        acc[0] += xv * w0.x;  acc[1] += xv * w0.y;
        acc[2] += xv * w0.z;  acc[3] += xv * w0.w;
        acc[4] += xv * w1v.x; acc[5] += xv * w1v.y;
        acc[6] += xv * w1v.z; acc[7] += xv * w1v.w;
    }
#pragma unroll
    for (int e = 0; e < E_; ++e)
        for (int off = 16; off; off >>= 1)
            acc[e] += __shfl_xor(acc[e], off, 32);
    if (lane == 0) {
        float m1 = -1e30f, m2 = -1e30f; int i1 = 0;
#pragma unroll
        for (int e = 0; e < E_; ++e) {
            const float l = acc[e] + br[e];
            if (l > m1) { m2 = m1; m1 = l; i1 = e; }
            else if (l > m2) { m2 = l; }
        }
        gate[t] = 1.f / (1.f + expf(m2 - m1));   // p0/(p0+p1)
        top1[t] = i1;
        rank[t] = atomicAdd(&cnt[i1], 1);
    }
}

// ---------------------------------------------------------------- scan + perm
__global__ __launch_bounds__(32) void moe_scan(const int* __restrict__ cnt,
                                               int* __restrict__ basep) {
    if (threadIdx.x == 0) {
        int s = 0;
        for (int e = 0; e < E_; ++e) { basep[e] = s; s += cnt[e]; }
        basep[E_] = s;
    }
}

__global__ __launch_bounds__(256) void moe_perm(const int* __restrict__ top1,
                                                const int* __restrict__ rank,
                                                const int* __restrict__ basep,
                                                int* __restrict__ perm) {
    const int t = blockIdx.x * 256 + threadIdx.x;
    perm[basep[top1[t]] + rank[t]] = t;
}

// ---------------------------------------------------------------- FFN1
// hmid[slot, i] = gelu( xh[perm[slot]] @ w1[e] + b1[e] ); A-tile via
// global_load_async_to_lds_b128 (per-lane gathered rows).
__global__ __launch_bounds__(256) void moe_ffn1(const _Float16* __restrict__ xh,
                                                const float* __restrict__ w1,
                                                const float* __restrict__ b1,
                                                const int* __restrict__ perm,
                                                const int* __restrict__ basep,
                                                const int* __restrict__ cnt,
                                                _Float16* __restrict__ hmid) {
    const int e  = blockIdx.z;
    const int mt = blockIdx.y;
    const int mcnt = cnt[e];
    if (mt * 64 >= mcnt) return;
    const int n0 = blockIdx.x * 128;
    const int m0 = basep[e] + mt * 64;

    __shared__ _Float16 lA[64][40];
    __shared__ _Float16 lB[128][40];

    const int tid  = threadIdx.x;
    const int lane = tid & 31, wave = tid >> 5;
    const int mw = wave & 3, nw = wave >> 2;
    const int hi  = (lane >> 4) & 1;
    const int l15 = lane & 15;

    v8f acc[4] = {};

    // A-fill mapping: 4 threads/row, 16B each. Rows past mcnt are clamped to
    // the last valid token (their outputs are masked in the epilogue).
    const int arow = tid >> 2;
    const int aseg = (tid & 3) * 8;
    int ridx = mt * 64 + arow;
    if (ridx >= mcnt) ridx = mcnt - 1;
    const int tok = perm[basep[e] + ridx];
    const unsigned lA_off = (unsigned)(size_t)&lA[arow][aseg];

    for (int k0 = 0; k0 < H_; k0 += 32) {
        __syncthreads();
        // ---- A tile: async gathered copy, 64 rows x 32 f16
        async_g2l_b128(lA_off, xh + (size_t)tok * H_ + k0 + aseg);
        // ---- B tile: 32(k) x 128(n) f32 -> f16, transposed [n][k], packed
        //      K-pairs so each store is one ds_store_b32.
#pragma unroll
        for (int r = 0; r < 2; ++r) {
            const int item = tid + r * 256;          // 0..511
            const int kk = (item >> 5) << 1;         // 0,2,..,30
            const int c4 = (item & 31) << 2;         // 0..124
            const float* wp = w1 + ((size_t)e * H_ + (k0 + kk)) * I_ + n0 + c4;
            const float4 a0 = *(const float4*)wp;
            const float4 a1 = *(const float4*)(wp + I_);
            store_pair_f16(&lB[c4 + 0][kk], a0.x, a1.x);
            store_pair_f16(&lB[c4 + 1][kk], a0.y, a1.y);
            store_pair_f16(&lB[c4 + 2][kk], a0.z, a1.z);
            store_pair_f16(&lB[c4 + 3][kk], a0.w, a1.w);
        }
        wait_async0();
        __syncthreads();
        // ---- WMMA
        const int ar = mw * 16 + l15;
        const int abase = hi ? 16 : 0;
        const v8h alo = *(const v8h*)&lA[ar][abase];
        const v8h ahi = *(const v8h*)&lA[ar][abase + 8];
        const v16h afrag = __builtin_shufflevector(alo, ahi,
            0, 1, 2, 3, 4, 5, 6, 7, 8, 9, 10, 11, 12, 13, 14, 15);
        const int kb = hi ? 16 : 0;
#pragma unroll
        for (int nn = 0; nn < 4; ++nn) {
            const int c = nw * 64 + nn * 16 + l15;
            const v16h bfrag = *(const v16h*)&lB[c][kb];
            acc[nn] = __builtin_amdgcn_wmma_f32_16x16x32_f16(
                false, afrag, false, bfrag, (short)0, acc[nn], false, false);
        }
    }
    // ---- epilogue: bias + exact GELU -> f16 hmid (slot-indexed)
    const int rofs = hi ? 8 : 0;
#pragma unroll
    for (int nn = 0; nn < 4; ++nn) {
        const int ncol = n0 + nw * 64 + nn * 16 + l15;
        const float bias = b1[e * I_ + ncol];
#pragma unroll
        for (int v = 0; v < 8; ++v) {
            const int rr = mw * 16 + rofs + v;
            if (mt * 64 + rr < mcnt) {
                float g = acc[nn][v] + bias;
                g = 0.5f * g * (1.f + erff(g * 0.70710678118f));
                hmid[(size_t)(m0 + rr) * I_ + ncol] = (_Float16)g;
            }
        }
    }
}

// ---------------------------------------------------------------- FFN2
// out[perm[slot], h] = ( hmid[slot] @ w2[e] + b2[e] ) * gate[perm[slot]];
// A-tile via TDM tensor_load_to_lds (contiguous expert-grouped rows).
__global__ __launch_bounds__(256) void moe_ffn2(const _Float16* __restrict__ hmid,
                                                const float* __restrict__ w2,
                                                const float* __restrict__ b2,
                                                const int* __restrict__ perm,
                                                const int* __restrict__ basep,
                                                const int* __restrict__ cnt,
                                                const float* __restrict__ gate,
                                                float* __restrict__ out) {
    const int e  = blockIdx.z;
    const int mt = blockIdx.y;
    const int mcnt = cnt[e];
    if (mt * 64 >= mcnt) return;
    const int n0 = blockIdx.x * 128;
    const int m0 = basep[e] + mt * 64;
    const unsigned rows_rem = (unsigned)(mcnt - mt * 64);

    __shared__ _Float16 lA[64][40];
    __shared__ _Float16 lB[128][40];

    const int tid  = threadIdx.x;
    const int lane = tid & 31, wave = tid >> 5;
    const int mw = wave & 3, nw = wave >> 2;
    const int hi  = (lane >> 4) & 1;
    const int l15 = lane & 15;

    v8f acc[4] = {};
    const unsigned lA_base = (unsigned)(size_t)&lA[0][0];

    for (int k0 = 0; k0 < I_; k0 += 32) {
        __syncthreads();
        // ---- A tile: one TDM DMA per block per K-step (wave 0 issues)
        if (wave == 0) {
            tdm_load_tile_f16(lA_base, hmid + (size_t)m0 * I_ + k0, rows_rem);
        }
        // ---- B tile: w2[e] is I_ x H_, f32 -> f16 transposed [n][k], packed
#pragma unroll
        for (int r = 0; r < 2; ++r) {
            const int item = tid + r * 256;
            const int kk = (item >> 5) << 1;
            const int c4 = (item & 31) << 2;
            const float* wp = w2 + ((size_t)e * I_ + (k0 + kk)) * H_ + n0 + c4;
            const float4 a0 = *(const float4*)wp;
            const float4 a1 = *(const float4*)(wp + H_);
            store_pair_f16(&lB[c4 + 0][kk], a0.x, a1.x);
            store_pair_f16(&lB[c4 + 1][kk], a0.y, a1.y);
            store_pair_f16(&lB[c4 + 2][kk], a0.z, a1.z);
            store_pair_f16(&lB[c4 + 3][kk], a0.w, a1.w);
        }
        if (wave == 0) __builtin_amdgcn_s_wait_tensorcnt(0);
        __syncthreads();
        const int ar = mw * 16 + l15;
        const int abase = hi ? 16 : 0;
        const v8h alo = *(const v8h*)&lA[ar][abase];
        const v8h ahi = *(const v8h*)&lA[ar][abase + 8];
        const v16h afrag = __builtin_shufflevector(alo, ahi,
            0, 1, 2, 3, 4, 5, 6, 7, 8, 9, 10, 11, 12, 13, 14, 15);
        const int kb = hi ? 16 : 0;
#pragma unroll
        for (int nn = 0; nn < 4; ++nn) {
            const int c = nw * 64 + nn * 16 + l15;
            const v16h bfrag = *(const v16h*)&lB[c][kb];
            acc[nn] = __builtin_amdgcn_wmma_f32_16x16x32_f16(
                false, afrag, false, bfrag, (short)0, acc[nn], false, false);
        }
    }
    // ---- epilogue: bias, gate scale, scatter to original token row (f32)
    const int rofs = hi ? 8 : 0;
#pragma unroll
    for (int v = 0; v < 8; ++v) {
        const int rr = mw * 16 + rofs + v;
        if (mt * 64 + rr < mcnt) {
            const int tokv = perm[m0 + rr];
            const float gt = gate[tokv];
#pragma unroll
            for (int nn = 0; nn < 4; ++nn) {
                const int ncol = n0 + nw * 64 + nn * 16 + l15;
                out[(size_t)tokv * H_ + ncol] = (acc[nn][v] + b2[e * H_ + ncol]) * gt;
            }
        }
    }
}

// ---------------------------------------------------------------- launch
extern "C" void kernel_launch(void* const* d_in, const int* in_sizes, int n_in,
                              void* d_out, int out_size, void* d_ws, size_t ws_size,
                              hipStream_t stream) {
    (void)in_sizes; (void)n_in; (void)out_size; (void)ws_size;
    const float* x  = (const float*)d_in[0];   // (B,S,H)
    const float* w1 = (const float*)d_in[1];   // (E,H,I)
    const float* b1 = (const float*)d_in[2];   // (E,I)
    const float* w2 = (const float*)d_in[3];   // (E,I,H)
    const float* b2 = (const float*)d_in[4];   // (E,H)
    const float* wr = (const float*)d_in[5];   // (H,E)
    const float* br = (const float*)d_in[6];   // (E,)
    float* out = (float*)d_out;

    // workspace layout (~84 MB total)
    char* p = (char*)d_ws;
    _Float16* xh   = (_Float16*)p; p += (size_t)T_ * H_ * sizeof(_Float16);  // 16 MB
    _Float16* hmid = (_Float16*)p; p += (size_t)T_ * I_ * sizeof(_Float16);  // 64 MB
    float* gate = (float*)p;   p += (size_t)T_ * sizeof(float);
    int* top1   = (int*)p;     p += (size_t)T_ * sizeof(int);
    int* rank   = (int*)p;     p += (size_t)T_ * sizeof(int);
    int* perm   = (int*)p;     p += (size_t)T_ * sizeof(int);
    int* cnt    = (int*)p;     p += 16 * sizeof(int);
    int* basep  = (int*)p;     p += 16 * sizeof(int);

    moe_init<<<1, 32, 0, stream>>>(cnt);
    moe_router<<<T_ / 8, 256, 0, stream>>>(x, wr, br, xh, top1, rank, gate, cnt);
    moe_scan<<<1, 32, 0, stream>>>(cnt, basep);
    moe_perm<<<T_ / 256, 256, 0, stream>>>(top1, rank, basep, perm);
    moe_ffn1<<<dim3(I_ / 128, T_ / 64, E_), 256, 0, stream>>>(xh, w1, b1, perm, basep, cnt, hmid);
    moe_ffn2<<<dim3(H_ / 128, T_ / 64, E_), 256, 0, stream>>>(hmid, w2, b2, perm, basep, cnt, gate, out);
}
